// QLSTM_65481071403230
// MI455X (gfx1250) — matrix operand order repaired and checked by reading, not compiled, not run on hardware
//
#include <hip/hip_runtime.h>
#include <hip/hip_bf16.h>
#include <math.h>

#define SEQ    128
#define BATCH  1024
#define IN_DIM 256
#define NQ     8
#define NG     4
#define NROW   32            // 4 gates * 8 outputs
#define MTOT   (SEQ*BATCH)   // 131072 rows

typedef _Float16 v16h __attribute__((ext_vector_type(16)));
typedef float    v8f  __attribute__((ext_vector_type(8)));

// ---------------- phase 0: weight prep ----------------
// Builds: wx16[32][256] f16 (gate-major input weights), whh[32][8] f32 (recurrent),
// ba[32] = bias + RX-param (folded into cos argument), cb[32] = cos(RY-param).
__global__ void qlstm_prep(
    const float* __restrict__ Wf, const float* __restrict__ bf, const float* __restrict__ pf,
    const float* __restrict__ Wi, const float* __restrict__ bi, const float* __restrict__ pi,
    const float* __restrict__ Wg, const float* __restrict__ bg, const float* __restrict__ pg,
    const float* __restrict__ Wo, const float* __restrict__ bo, const float* __restrict__ po,
    _Float16* __restrict__ wx16, float* __restrict__ whh,
    float* __restrict__ ba, float* __restrict__ cb)
{
  const float* Ws[NG] = {Wf, Wi, Wg, Wo};
  const float* bs[NG] = {bf, bi, bg, bo};
  const float* ps[NG] = {pf, pi, pg, po};
  const int fan = IN_DIM + NQ;
  int tid = threadIdx.x;
  for (int idx = tid; idx < NROW * IN_DIM; idx += blockDim.x) {
    int n32 = idx >> 8, k = idx & 255;
    wx16[idx] = (_Float16)Ws[n32 >> 3][(n32 & 7) * fan + k];
  }
  for (int idx = tid; idx < NROW * NQ; idx += blockDim.x) {
    int n32 = idx >> 3, j = idx & 7;
    whh[idx] = Ws[n32 >> 3][(n32 & 7) * fan + IN_DIM + j];
  }
  if (tid < NROW) {
    int g = tid >> 3, n = tid & 7;
    ba[tid] = bs[g][n] + ps[g][n * 3 + 0];   // bias + alpha folded
    cb[tid] = cosf(ps[g][n * 3 + 1]);        // cos(beta)
  }
}

// ---------------- phase 1: input projection GEMM (WMMA) ----------------
// preX[r][0:32] = X[r][0:256] @ Wx.T  for all r in [0, SEQ*BATCH).
// One wave = one 16-row M tile x full N=32 (two 16x16 f32 accumulators).
__global__ void qlstm_gemm(const float* __restrict__ X,
                           const _Float16* __restrict__ W16,
                           float* __restrict__ preX)
{
  int wave = (int)((blockIdx.x * blockDim.x + threadIdx.x) >> 5);
  int lane = threadIdx.x & 31;
  int l15 = lane & 15, hi = lane >> 4;

  const float*    xrow = X   + ((size_t)wave * 16 + l15) * IN_DIM;   // A row m=l15
  const _Float16* w0   = W16 + (size_t)l15        * IN_DIM;          // B col n=l15
  const _Float16* w1   = W16 + (size_t)(16 + l15) * IN_DIM;          // B col n=16+l15

  v8f c0 = {};
  v8f c1 = {};

  #pragma unroll
  for (int k0 = 0; k0 < IN_DIM; k0 += 32) {
    // A fragment (K contiguous per lane-half, matching B): lanes<16 hold K 0-15;
    // lanes>=16 hold K 16-31 -> two contiguous 32B f32 chunks, cvt to f16.
    const float4* pa = (const float4*)(xrow + k0 + 16 * hi);
    float4 a0 = pa[0], a1 = pa[1];
    const float4* pb = (const float4*)(xrow + k0 + 16 * hi + 8);
    float4 a2 = pb[0], a3 = pb[1];
    v16h A;
    A[0] = (_Float16)a0.x; A[1] = (_Float16)a0.y; A[2]  = (_Float16)a0.z; A[3]  = (_Float16)a0.w;
    A[4] = (_Float16)a1.x; A[5] = (_Float16)a1.y; A[6]  = (_Float16)a1.z; A[7]  = (_Float16)a1.w;
    A[8] = (_Float16)a2.x; A[9] = (_Float16)a2.y; A[10] = (_Float16)a2.z; A[11] = (_Float16)a2.w;
    A[12]= (_Float16)a3.x; A[13]= (_Float16)a3.y; A[14] = (_Float16)a3.z; A[15] = (_Float16)a3.w;

    // B fragment (lane = column N, K contiguous per lane-half): 16 contiguous halves.
    v16h B0 = *(const v16h*)(w0 + k0 + 16 * hi);
    v16h B1 = *(const v16h*)(w1 + k0 + 16 * hi);

    c0 = __builtin_amdgcn_wmma_f32_16x16x32_f16(false, A, false, B0, (short)0, c0, false, false);
    c1 = __builtin_amdgcn_wmma_f32_16x16x32_f16(false, A, false, B1, (short)0, c1, false, false);
  }

  // C/D layout: VGPR r, lane L -> (m = r + 8*(L>=16), n = L&15)
  float* orow = preX + (size_t)wave * 16 * NROW;
  #pragma unroll
  for (int r = 0; r < 8; ++r) {
    int m = r + 8 * hi;
    orow[(size_t)m * NROW + l15]      = c0[r];
    orow[(size_t)m * NROW + 16 + l15] = c1[r];
  }
}

// ---------------- phase 2: sequential LSTM recurrence ----------------
__device__ __forceinline__ float fast_sigmoid(float x) { return 1.0f / (1.0f + __expf(-x)); }
__device__ __forceinline__ float fast_tanh(float x) {
  float e = __expf(2.0f * x);
  return (e - 1.0f) / (e + 1.0f);   // |x| <= ~4 here, no overflow
}

__global__ void qlstm_recur(const float* __restrict__ preX,
                            const float* __restrict__ whh,
                            const float* __restrict__ ba,
                            const float* __restrict__ cb,
                            float* __restrict__ out)
{
  int b = blockIdx.x * blockDim.x + threadIdx.x;   // one batch element per lane
  float hx[NQ], cx[NQ];
  #pragma unroll
  for (int k = 0; k < NQ; ++k) { hx[k] = 0.f; cx[k] = 0.f; }

  for (int t = 0; t < SEQ; ++t) {
    const float* pp = preX + ((size_t)t * BATCH + b) * NROW;
    float pre[NROW];
    #pragma unroll
    for (int i = 0; i < NROW / 4; ++i) {
      float4 v = ((const float4*)pp)[i];
      pre[4 * i] = v.x; pre[4 * i + 1] = v.y; pre[4 * i + 2] = v.z; pre[4 * i + 3] = v.w;
    }
    if (t + 1 < SEQ)  // global_prefetch_b8 for next timestep's pre-activations
      __builtin_prefetch(preX + ((size_t)(t + 1) * BATCH + b) * NROW, 0, 0);

    float act[NG][NQ];
    #pragma unroll
    for (int g = 0; g < NG; ++g) {
      // z_i = cos(beta_i) * cos(angle_i + alpha_i); angle = pre + bias + Whh@hx
      float z[NQ];
      #pragma unroll
      for (int k = 0; k < NQ; ++k) {
        int n = g * NQ + k;
        float ang = pre[n] + ba[n];
        #pragma unroll
        for (int j = 0; j < NQ; ++j) ang = fmaf(whh[n * NQ + j], hx[j], ang);
        z[k] = cb[n] * __cosf(ang);
      }
      // CNOT ring in Heisenberg picture: e_0 = z1..z7, e_k = z0..zk (k>=1)
      float e[NQ];
      e[0] = z[1] * z[2] * z[3] * z[4] * z[5] * z[6] * z[7];
      e[1] = z[0] * z[1];
      #pragma unroll
      for (int k = 2; k < NQ; ++k) e[k] = e[k - 1] * z[k];
      #pragma unroll
      for (int k = 0; k < NQ; ++k)
        act[g][k] = (g == 2) ? fast_tanh(e[k]) : fast_sigmoid(e[k]);
    }

    #pragma unroll
    for (int k = 0; k < NQ; ++k) {
      float c2 = act[0][k] * cx[k] + act[1][k] * act[2][k];
      cx[k] = c2;
      hx[k] = act[3][k] * fast_tanh(c2);
    }

    float* op = out + ((size_t)t * BATCH + b) * NQ;
    float4 v0 = {hx[0], hx[1], hx[2], hx[3]};
    float4 v1 = {hx[4], hx[5], hx[6], hx[7]};
    ((float4*)op)[0] = v0;
    ((float4*)op)[1] = v1;
  }

  float* hp = out + (size_t)SEQ * BATCH * NQ + (size_t)b * NQ;
  float* cp = hp + (size_t)BATCH * NQ;
  float4 h0 = {hx[0], hx[1], hx[2], hx[3]}, h1 = {hx[4], hx[5], hx[6], hx[7]};
  float4 c0 = {cx[0], cx[1], cx[2], cx[3]}, c1 = {cx[4], cx[5], cx[6], cx[7]};
  ((float4*)hp)[0] = h0; ((float4*)hp)[1] = h1;
  ((float4*)cp)[0] = c0; ((float4*)cp)[1] = c1;
}

// ---------------- launcher ----------------
extern "C" void kernel_launch(void* const* d_in, const int* in_sizes, int n_in,
                              void* d_out, int out_size, void* d_ws, size_t ws_size,
                              hipStream_t stream) {
  // setup_inputs() dict order: inputs, Wf,bf,pf, Wi,bi,pi, Wg,bg,pg, Wo,bo,po
  const float* X  = (const float*)d_in[0];
  const float* Wf = (const float*)d_in[1];
  const float* bf = (const float*)d_in[2];
  const float* pf = (const float*)d_in[3];
  const float* Wi = (const float*)d_in[4];
  const float* bi = (const float*)d_in[5];
  const float* pi = (const float*)d_in[6];
  const float* Wg = (const float*)d_in[7];
  const float* bg = (const float*)d_in[8];
  const float* pg = (const float*)d_in[9];
  const float* Wo = (const float*)d_in[10];
  const float* bo = (const float*)d_in[11];
  const float* po = (const float*)d_in[12];

  char* ws = (char*)d_ws;
  _Float16* wx16 = (_Float16*)(ws);          // 32*256 f16  = 16384 B
  float*    whh  = (float*)(ws + 16384);     // 32*8  f32   =  1024 B
  float*    ba   = (float*)(ws + 17408);     // 32    f32   =   128 B
  float*    cb   = (float*)(ws + 17536);     // 32    f32   =   128 B
  float*    preX = (float*)(ws + 18432);     // 131072*32 f32 = 16 MB
  float*    out  = (float*)d_out;

  qlstm_prep <<<1, 256, 0, stream>>>(Wf, bf, pf, Wi, bi, pi, Wg, bg, pg, Wo, bo, po,
                                     wx16, whh, ba, cb);
  qlstm_gemm <<<MTOT / 16 / 8, 256, 0, stream>>>(X, wx16, preX);   // 8192 waves, 1 tile each
  qlstm_recur<<<BATCH / 64, 64, 0, stream>>>(preX, whh, ba, cb, out);
}
